// MultiHeadAttention_72576357368165
// MI455X (gfx1250) — matrix operand order repaired and checked by reading, not compiled, not run on hardware
//
#include <hip/hip_runtime.h>
#include <hip/hip_bf16.h>
#include <math.h>

// ---------------------------------------------------------------------------
// MHA (QKV gemm + RoPE, flash attention, out projection) for gfx1250 (MI455X).
// bf16 WMMA (v_wmma_f32_16x16x32_bf16), f32 accumulate. Inputs pre-converted
// to bf16 once (memory-bound, ~2us at 23.3 TB/s) so GEMM loops are pure
// load+wmma. QKV GEMM stages weight tiles into LDS via the Tensor Data Mover
// (tensor_load_to_lds, double buffered, s_wait_tensorcnt). Attention computes
// S^T = K*Q^T so softmax rows are lane-local; 32 queries/wave for 2x reuse of
// K/V tiles.
// ---------------------------------------------------------------------------

#define B_SZ   2
#define T_SEQ  2048
#define DIMC   1024
#define H_NUM  16
#define HD     64
#define LOG2_THETA 13.287712379549449f  // log2(10000)

typedef __attribute__((ext_vector_type(16))) __bf16 v16bf;
typedef __attribute__((ext_vector_type(8)))  __bf16 bf8;
typedef __attribute__((ext_vector_type(8)))  float  v8f;
typedef __attribute__((ext_vector_type(4)))  float  f4;
typedef __attribute__((ext_vector_type(4)))  unsigned int v4u;
typedef __attribute__((ext_vector_type(8)))  int    v8i;
typedef __attribute__((ext_vector_type(4)))  int    v4i;

static __device__ inline v8f zero8() {
    v8f z;
#pragma unroll
    for (int i = 0; i < 8; ++i) z[i] = 0.0f;
    return z;
}

static __device__ inline v8f wmma_bf16(v16bf a, v16bf b, v8f c) {
    // (neg_a, A, neg_b, B, c_mod, C, reuse_a, reuse_b)
    return __builtin_amdgcn_wmma_f32_16x16x32_bf16(false, a, false, b,
                                                   (short)0, c, false, false);
}

// 16x32 A tile from bf16 memory: lane's 16 contiguous K at p[0..15].
static __device__ inline v16bf a_tile_from_bf16(const __bf16* p) {
    bf8 lo = *(const bf8*)p;
    bf8 hi = *(const bf8*)(p + 8);
    v16bf a;
#pragma unroll
    for (int i = 0; i < 8; ++i) { a[i] = lo[i]; a[8 + i] = hi[i]; }
    return a;
}

// 32x16 B tile: 16 contiguous bf16 per lane (works for global or LDS ptrs).
static __device__ inline v16bf b_tile_from_bf16(const __bf16* p) {
    bf8 lo = *(const bf8*)p;
    bf8 hi = *(const bf8*)(p + 8);
    v16bf b;
#pragma unroll
    for (int i = 0; i < 8; ++i) { b[i] = lo[i]; b[8 + i] = hi[i]; }
    return b;
}

// ---------------------------------------------------------------------------
// TDM: load a 64-row x 32-col bf16 tile (row stride 1024 elems) into LDS.
// D# per CDNA5 ISA ch.8: group0 = {count/lds_addr/global_addr/type=2},
// group1 = {data_size=1(2B), tensor dims, tile dims, dim0 stride}. Groups 2/3
// are zero (2D tensor: tile_dim2 == 0 -> unused).
// ---------------------------------------------------------------------------
static __device__ inline void tdm_load_tile_64x32(unsigned lds_off,
                                                  unsigned long long ga,
                                                  unsigned tensor_rows) {
    v4u g0 = { 1u,                                  // count = 1 valid D#
               lds_off,                             // lds_addr (bytes)
               (unsigned)(ga & 0xffffffffu),        // global_addr[31:0]
               (unsigned)(ga >> 32) | 0x80000000u };// ga[56:32] | type=2<<30
    v8i g1 = { (int)(1u << 16),                     // data_size = 1 (2 bytes)
               (int)(1024u << 16),                  // tensor_dim0 = 1024
               (int)((tensor_rows & 0xffffu) << 16),// tensor_dim1 low16
               (int)(32u << 16),                    // tile_dim0 = 32 (dim1 hi=0)
               64,                                  // tile_dim1 = 64
               1024,                                // tensor_dim0_stride
               0, 0 };
    v4i g2 = { 0, 0, 0, 0 };
    v4i g3 = { 0, 0, 0, 0 };
#if defined(__clang_major__) && (__clang_major__ >= 23)
    v8i gx = { 0, 0, 0, 0, 0, 0, 0, 0 };
    __builtin_amdgcn_tensor_load_to_lds(g0, g1, g2, g3, gx, 0);
#else
    __builtin_amdgcn_tensor_load_to_lds(g0, g1, g2, g3, 0);
#endif
}

// ---------------------------------------------------------------------------
// Kernel 0: bulk f32 -> bf16 convert (8 elems/thread, b128 in, b128 out).
// ---------------------------------------------------------------------------
__global__ void __launch_bounds__(256)
cvt_bf16_kernel(const float* __restrict__ src, __bf16* __restrict__ dst, int n8)
{
    int i = blockIdx.x * blockDim.x + threadIdx.x;
    if (i >= n8) return;
    f4 a = ((const f4*)src)[2 * i];
    f4 b = ((const f4*)src)[2 * i + 1];
    bf8 o;
#pragma unroll
    for (int j = 0; j < 4; ++j) { o[j] = (__bf16)a[j]; o[4 + j] = (__bf16)b[j]; }
    ((bf8*)dst)[i] = o;
}

// ---------------------------------------------------------------------------
// Kernel 1: qkv = x @ w_qkv^T with fused RoPE + layout transform.
// Block = 128 threads (4 waves) covering 64 tokens x 64 features. The 64x32
// weight sub-block per K-step is staged into LDS by the TDM (double
// buffered); each wave consumes it for its own 16-token M-subtile.
//   q,k -> bf16 [B,H,T,D] (RoPE applied), v -> bf16 [B,H,D,T] (transposed).
// ---------------------------------------------------------------------------
__global__ void __launch_bounds__(128)
qkv_rope_kernel(const __bf16* __restrict__ xb, const __bf16* __restrict__ wqb,
                __bf16* __restrict__ q, __bf16* __restrict__ k,
                __bf16* __restrict__ vt)
{
    __shared__ __bf16 wtile[2][64 * 32];   // 2 x 4KB double buffer

    const int tid  = threadIdx.x;
    const int warp = tid >> 5;
    const int lane = tid & 31;
    const int l15  = lane & 15;
    const int half = lane >> 4;
    const int nb   = blockIdx.x * 64;            // feature base [0,3072)
    const int m0   = blockIdx.y * 64 + warp * 16;// token-row base [0,4096)

    v8f c[4];
#pragma unroll
    for (int j = 0; j < 4; ++j) c[j] = zero8();

    const unsigned long long wga = (unsigned long long)(const void*)wqb;
    const unsigned lds0 = (unsigned)(size_t)(&wtile[0][0]);
    const unsigned lds1 = (unsigned)(size_t)(&wtile[1][0]);

    if (warp == 0)
        tdm_load_tile_64x32(lds0, wga + 2ull * ((size_t)nb * DIMC + 0), 3072u);

    const __bf16* arow = xb + (size_t)(m0 + l15) * DIMC;
    const int nit = DIMC / 32;
    for (int it = 0; it < nit; ++it) {
        const int k0 = it * 32;
        if (warp == 0) __builtin_amdgcn_s_wait_tensorcnt(0);
        __syncthreads();   // current buffer ready; prev buffer reads retired
        if (warp == 0 && it + 1 < nit)
            tdm_load_tile_64x32((it & 1) ? lds0 : lds1,
                                wga + 2ull * ((size_t)nb * DIMC + k0 + 32),
                                3072u);

        const __bf16* wl = &wtile[it & 1][0];
        v16bf a = a_tile_from_bf16(arow + k0 + half * 16);
#pragma unroll
        for (int j = 0; j < 4; ++j) {
            v16bf b = b_tile_from_bf16(wl + (j * 16 + l15) * 32 + half * 16);
            c[j] = wmma_bf16(a, b, c[j]);
        }
    }

    const int sec = nb / DIMC;           // 0=q, 1=k, 2=v
    const int h   = (nb % DIMC) / HD;

#pragma unroll
    for (int j = 0; j < 4; ++j) {
#pragma unroll
        for (int r = 0; r < 8; ++r) {
            const int nrow = m0 + r + 8 * half;
            const int b    = nrow >> 11;
            const int t    = nrow & (T_SEQ - 1);
            const int d    = j * 16 + l15;
            const float val = c[j][r];
            if (sec == 2) {
                vt[((size_t)(b * H_NUM + h) * HD + d) * T_SEQ + t] = (__bf16)val;
            } else {
                const int   dm  = d & 31;
                const float inv = exp2f(-(float)dm * (LOG2_THETA / 32.0f));
                const float ang = (float)t * inv;
                const float cs  = __cosf(ang);
                const float sn  = __sinf(ang);
                const float prt = c[j ^ 2][r];            // element at d ^ 32
                const float rot = (d < 32) ? -prt : prt;  // rotate_half
                const __bf16 ob = (__bf16)(val * cs + rot * sn);
                const size_t idx =
                    ((size_t)(b * H_NUM + h) * T_SEQ + t) * HD + d;
                if (sec == 0) q[idx] = ob; else k[idx] = ob;
            }
        }
    }
}

// ---------------------------------------------------------------------------
// Kernel 2: flash attention, one wave per (b, h, 32-query tile).
// S^T = K @ Q^T -> softmax rows live in VGPRs (lane-local, one xor-16
// shuffle to merge lane halves). out^T = V^T @ P^T. Every K/V tile is reused
// by both query sub-tiles: 16 WMMA per 16 b128 loads per 32-key chunk.
// ---------------------------------------------------------------------------
__global__ void __launch_bounds__(32)
attn_kernel(const __bf16* __restrict__ q, const __bf16* __restrict__ k,
            const __bf16* __restrict__ vt, __bf16* __restrict__ ao)
{
    const int lane = threadIdx.x;
    const int l15  = lane & 15;
    const int half = lane >> 4;
    const int t0   = blockIdx.x * 32;
    const int h    = blockIdx.y;
    const int b    = blockIdx.z;

    // Q^T B-tiles: bq[qt][dhalf], lane n holds column t = t0+qt*16+(n&15).
    v16bf bq[2][2];
#pragma unroll
    for (int qt = 0; qt < 2; ++qt) {
        const __bf16* qrow =
            q + ((size_t)(b * H_NUM + h) * T_SEQ + t0 + qt * 16 + l15) * HD;
        bq[qt][0] = b_tile_from_bf16(qrow + half * 16);
        bq[qt][1] = b_tile_from_bf16(qrow + 32 + half * 16);
    }

    const __bf16* kbase = k  + (size_t)(b * H_NUM + h) * T_SEQ * HD;
    const __bf16* vbase = vt + (size_t)(b * H_NUM + h) * HD * T_SEQ;

    v8f acc[2][4];
#pragma unroll
    for (int qt = 0; qt < 2; ++qt)
#pragma unroll
        for (int j = 0; j < 4; ++j) acc[qt][j] = zero8();
    float m[2] = { -1e30f, -1e30f }, lsum[2] = { 0.0f, 0.0f };

    for (int s0 = 0; s0 < T_SEQ; s0 += 32) {
        if (s0 + 32 < T_SEQ)  // warm caches for next K chunk
            __builtin_prefetch(kbase + (size_t)(s0 + 32 + l15) * HD, 0, 1);

        // ---- scores: S^T(32s x 16t) per query sub-tile ----
        v8f ca[2] = { zero8(), zero8() }, cb[2] = { zero8(), zero8() };
#pragma unroll
        for (int ssub = 0; ssub < 2; ++ssub)
#pragma unroll
            for (int dh = 0; dh < 2; ++dh) {
                v16bf a = a_tile_from_bf16(
                    kbase + (size_t)(s0 + ssub * 16 + l15) * HD + dh * 32 +
                    half * 16);
#pragma unroll
                for (int qt = 0; qt < 2; ++qt) {
                    if (ssub == 0) ca[qt] = wmma_bf16(a, bq[qt][dh], ca[qt]);
                    else           cb[qt] = wmma_bf16(a, bq[qt][dh], cb[qt]);
                }
            }

        // ---- online softmax + rebuild P^T as B tiles ----
        v16bf pb[2];
#pragma unroll
        for (int qt = 0; qt < 2; ++qt) {
            float p16[16];
            float cmax = -1e30f;
#pragma unroll
            for (int r = 0; r < 8; ++r) {
                p16[r]     = ca[qt][r] * 0.125f;   // 1/sqrt(64)
                p16[8 + r] = cb[qt][r] * 0.125f;
            }
#pragma unroll
            for (int i = 0; i < 16; ++i) cmax = fmaxf(cmax, p16[i]);
            cmax = fmaxf(cmax, __shfl_xor(cmax, 16));
            const float mn    = fmaxf(m[qt], cmax);
            const float alpha = __expf(m[qt] - mn);
            float rs = 0.0f;
#pragma unroll
            for (int i = 0; i < 16; ++i) {
                p16[i] = __expf(p16[i] - mn);
                rs += p16[i];
            }
            rs += __shfl_xor(rs, 16);
            lsum[qt] = lsum[qt] * alpha + rs;
            m[qt]    = mn;
#pragma unroll
            for (int j = 0; j < 4; ++j)
#pragma unroll
                for (int r = 0; r < 8; ++r) acc[qt][j][r] *= alpha;

            // half0 lanes hold s 0..15 (own Ca + partner Ca); half1 s 16..31.
            float xv[8], yv[8];
#pragma unroll
            for (int i = 0; i < 8; ++i) xv[i] = half ? p16[i] : p16[8 + i];
#pragma unroll
            for (int i = 0; i < 8; ++i) yv[i] = __shfl_xor(xv[i], 16);
#pragma unroll
            for (int i = 0; i < 8; ++i) {
                pb[qt][i]     = (__bf16)(half ? yv[i]      : p16[i]);
                pb[qt][8 + i] = (__bf16)(half ? p16[8 + i] : yv[i]);
            }
        }

        // ---- out^T += V^T @ P^T, 4 d-chunks, V tile shared by both qt ----
#pragma unroll
        for (int dt = 0; dt < 4; ++dt) {
            v16bf av = a_tile_from_bf16(
                vbase + (size_t)(dt * 16 + l15) * T_SEQ + s0 + half * 16);
#pragma unroll
            for (int qt = 0; qt < 2; ++qt)
                acc[qt][dt] = wmma_bf16(av, pb[qt], acc[qt][dt]);
        }
    }

#pragma unroll
    for (int qt = 0; qt < 2; ++qt) {
        const float invl = 1.0f / lsum[qt];
        const int   t    = t0 + qt * 16 + l15;
#pragma unroll
        for (int dt = 0; dt < 4; ++dt)
#pragma unroll
            for (int r = 0; r < 8; ++r) {
                const int d = dt * 16 + r + 8 * half;
                ao[(size_t)(b * T_SEQ + t) * DIMC + h * HD + d] =
                    (__bf16)(acc[qt][dt][r] * invl);
            }
    }
}

// ---------------------------------------------------------------------------
// Kernel 3: out = attn_out @ w_proj^T  (bf16 x bf16 -> f32 stores)
// ---------------------------------------------------------------------------
__global__ void __launch_bounds__(32)
proj_kernel(const __bf16* __restrict__ ao, const __bf16* __restrict__ wpb,
            float* __restrict__ out)
{
    const int lane = threadIdx.x;
    const int l15  = lane & 15;
    const int half = lane >> 4;
    const int nb   = blockIdx.x * 64;
    const int m0   = blockIdx.y * 16;

    v8f c[4];
#pragma unroll
    for (int j = 0; j < 4; ++j) c[j] = zero8();

    const __bf16* arow = ao + (size_t)(m0 + l15) * DIMC;
    for (int k0 = 0; k0 < DIMC; k0 += 32) {
        v16bf a = a_tile_from_bf16(arow + k0 + half * 16);
#pragma unroll
        for (int j = 0; j < 4; ++j) {
            v16bf b = b_tile_from_bf16(
                wpb + (size_t)(nb + j * 16 + l15) * DIMC + k0 + half * 16);
            c[j] = wmma_bf16(a, b, c[j]);
        }
    }

#pragma unroll
    for (int j = 0; j < 4; ++j)
#pragma unroll
        for (int r = 0; r < 8; ++r)
            out[(size_t)(m0 + r + 8 * half) * DIMC + nb + j * 16 + l15] = c[j][r];
}

// ---------------------------------------------------------------------------
extern "C" void kernel_launch(void* const* d_in, const int* in_sizes, int n_in,
                              void* d_out, int out_size, void* d_ws, size_t ws_size,
                              hipStream_t stream)
{
    (void)in_sizes; (void)n_in; (void)out_size; (void)ws_size;
    const float* x      = (const float*)d_in[0];
    const float* w_qkv  = (const float*)d_in[1];
    const float* w_proj = (const float*)d_in[2];
    float*       out    = (float*)d_out;

    const size_t QKV_ELEMS = (size_t)B_SZ * H_NUM * T_SEQ * HD;  // 4 Mi
    const size_t X_ELEMS   = (size_t)B_SZ * T_SEQ * DIMC;        // 4 Mi
    const size_t WQ_ELEMS  = (size_t)3 * DIMC * DIMC;            // 3 Mi
    const size_t WP_ELEMS  = (size_t)DIMC * DIMC;                // 1 Mi

    __bf16* q   = (__bf16*)d_ws;
    __bf16* k   = q   + QKV_ELEMS;
    __bf16* vt  = k   + QKV_ELEMS;
    __bf16* ao  = vt  + QKV_ELEMS;
    __bf16* xb  = ao  + QKV_ELEMS;
    __bf16* wqb = xb  + X_ELEMS;
    __bf16* wpb = wqb + WQ_ELEMS;

    // one-shot bf16 conversions (memory-bound)
    cvt_bf16_kernel<<<dim3((unsigned)(X_ELEMS / 8 + 255) / 256), 256, 0, stream>>>(
        x, xb, (int)(X_ELEMS / 8));
    cvt_bf16_kernel<<<dim3((unsigned)(WQ_ELEMS / 8 + 255) / 256), 256, 0, stream>>>(
        w_qkv, wqb, (int)(WQ_ELEMS / 8));
    cvt_bf16_kernel<<<dim3((unsigned)(WP_ELEMS / 8 + 255) / 256), 256, 0, stream>>>(
        w_proj, wpb, (int)(WP_ELEMS / 8));

    // qkv gemm + rope: 48 feature tiles x 64 token blocks, 4 waves each
    qkv_rope_kernel<<<dim3(48, 64), 128, 0, stream>>>(xb, wqb, q, k, vt);
    // attention: (T/32, H, B) waves, 32 queries per wave
    attn_kernel<<<dim3(T_SEQ / 32, H_NUM, B_SZ), 32, 0, stream>>>(q, k, vt, ao);
    // projection: 16 feature tiles x 256 token tiles
    proj_kernel<<<dim3(16, 256), 32, 0, stream>>>(ao, wpb, out);
}
